// AttnDecoder_35175782154802
// MI455X (gfx1250) — matrix operand order repaired and checked by reading, not compiled, not run on hardware
//
#include <hip/hip_runtime.h>
#include <hip/hip_bf16.h>
#include <cstdint>

#define B_  8
#define L_  128
#define H_  512
#define V_  32000
#define BL_ 1024   // B_*L_

typedef __attribute__((ext_vector_type(16))) __bf16 v16bf;
typedef __attribute__((ext_vector_type(8)))  __bf16 v8bf;
typedef __attribute__((ext_vector_type(8)))  float  v8f;

// ---------------------------------------------------------------------------
// WMMA fragment loaders (bf16 16x16x32, wave32)
// A layout (ISA 7.12.2, 16-bit A 16x32): lane l: m=l&15, khalf=l>>4,
//   element e -> K = 16*(e>>3) + 8*khalf + (e&7)  => two contiguous 8-elt runs.
// B layout (32x16, K-major per n): lane l: n=l&15, element e -> K = 16*(e>>3) + 8*(l>>4) + (e&7)
//   => two contiguous 8-elt runs from row-major [N][K] storage (same as A).
// ---------------------------------------------------------------------------
__device__ __forceinline__ v16bf load_a_frag(const __bf16* A, int lda, int m0, int k0, int lane){
  int m  = m0 + (lane & 15);
  int kh = lane >> 4;
  const __bf16* p0 = A + (long)m * lda + k0 + 8 * kh;
  union { v16bf v; v8bf h[2]; } u;
  u.h[0] = *(const v8bf*)p0;
  u.h[1] = *(const v8bf*)(p0 + 16);
  return u.v;
}

__device__ __forceinline__ v16bf load_b_frag(const __bf16* Bm, int ldb, int n0, int k0, int lane){
  int n  = n0 + (lane & 15);
  int kh = lane >> 4;
  const __bf16* p0 = Bm + (long)n * ldb + k0 + 8 * kh;
  union { v16bf v; v8bf h[2]; } u;
  u.h[0] = *(const v8bf*)p0;
  u.h[1] = *(const v8bf*)(p0 + 16);
  return u.v;
}

__device__ __forceinline__ float sigm(float x){ return 1.f / (1.f + expf(-x)); }

#define WMMA_BF16(A_, B_x, C_) \
  __builtin_amdgcn_wmma_f32_16x16x32_bf16(false, (A_), false, (B_x), (short)0, (C_), false, false)

// ---------------------------------------------------------------------------
// f32 -> bf16 conversion
// ---------------------------------------------------------------------------
__global__ void k_cvt(const float* __restrict__ s, __bf16* __restrict__ d, long n){
  long i  = (long)blockIdx.x * blockDim.x + threadIdx.x;
  long st = (long)gridDim.x * blockDim.x;
  for (; i < n; i += st) d[i] = (__bf16)s[i];
}

// embedded gather: f32 buffer + bf16 into xcat[:, 0:512]
__global__ void k_embed(const int* __restrict__ idx, const float* __restrict__ emb,
                        float* __restrict__ e32, __bf16* __restrict__ xcat){
  int row = blockIdx.x;                    // b*L + i
  int t   = idx[row];
  const float* src = emb + (long)t * H_;
  for (int h = threadIdx.x; h < H_; h += blockDim.x){
    float v = src[h];
    e32[(long)row * H_ + h]   = v;
    xcat[(long)row * 1024 + h] = (__bf16)v;
  }
}

// encoder_outputs (B,L,H) -> eoT_bf (B,H,L) bf16  (B-matrix for attn/context GEMMs)
__global__ void k_eoT(const float* __restrict__ eo, __bf16* __restrict__ eoT){
  long n = (long)B_ * H_ * L_;
  long i = (long)blockIdx.x * blockDim.x + threadIdx.x;
  long st = (long)gridDim.x * blockDim.x;
  for (; i < n; i += st){
    int b = (int)(i >> 16);          // H_*L_ = 65536
    int r = (int)(i & 65535);
    int h = r >> 7;
    int j = r & 127;
    eoT[i] = (__bf16)eo[(((long)b << 7) + j) * H_ + h];
  }
}

// hidterm[b*128+j] = dot(h0[b,:], attn_W[j, H:2H]) + attn_b[j]   (warp per output)
__global__ void k_hidterm(const float* __restrict__ h0, const float* __restrict__ attnW,
                          const float* __restrict__ attnb, float* __restrict__ out){
  int w    = blockIdx.x * (blockDim.x >> 5) + (threadIdx.x >> 5);
  int lane = threadIdx.x & 31;
  int b = w >> 7, j = w & 127;
  const float* hv = h0 + (long)b * H_;
  const float* wr = attnW + (long)j * 1024 + H_;
  float s = 0.f;
  for (int k = lane; k < H_; k += 32) s += hv[k] * wr[k];
  for (int off = 16; off; off >>= 1) s += __shfl_down(s, off, 32);
  if (!lane) out[w] = s + attnb[j];
}

// Generic bf16 WMMA GEMM: C[M,N] = A[M,K] @ B[N,K]^T (+bias) (optional relu)
// 2x2 register-blocked: each wave computes a 32x32 output patch (4 wmma per
// 4 fragment loads -> 2x arithmetic intensity vs 1 tile/wave).
// 8 waves/block span 256 N-columns; batched via blockIdx.z.
// Requires M%32==0, N%32==0, K%32==0 (true for all call sites here).
__global__ void __launch_bounds__(256)
k_gemm(const __bf16* __restrict__ A, int lda, long sA,
       const __bf16* __restrict__ Bm, int ldb, long sB,
       float* Cf, __bf16* Cb, int ldc, long sC,
       const float* __restrict__ bias, int biasMode,   // 0:none 1:per-col 2:bias[(row>>7)*N+col]
       int N, int K, int doRelu)
{
  int bz = blockIdx.z;
  A  += sA * bz;
  Bm += sB * bz;
  if (Cf) Cf += sC * bz;
  if (Cb) Cb += sC * bz;

  int wave = threadIdx.x >> 5, lane = threadIdx.x & 31;
  int n0 = (blockIdx.x * 8 + wave) * 32;
  if (n0 >= N) return;
  int m0 = blockIdx.y * 32;

  v8f c00 = {}, c01 = {}, c10 = {}, c11 = {};
  for (int k0 = 0; k0 < K; k0 += 32){
    v16bf a0 = load_a_frag(A,  lda, m0,      k0, lane);
    v16bf a1 = load_a_frag(A,  lda, m0 + 16, k0, lane);
    v16bf b0 = load_b_frag(Bm, ldb, n0,      k0, lane);
    v16bf b1 = load_b_frag(Bm, ldb, n0 + 16, k0, lane);
    c00 = WMMA_BF16(a0, b0, c00);
    c01 = WMMA_BF16(a0, b1, c01);
    c10 = WMMA_BF16(a1, b0, c10);
    c11 = WMMA_BF16(a1, b1, c11);
  }

#pragma unroll
  for (int mi = 0; mi < 2; mi++){
#pragma unroll
    for (int ni = 0; ni < 2; ni++){
      v8f acc = (mi == 0) ? (ni == 0 ? c00 : c01) : (ni == 0 ? c10 : c11);
      int col   = n0 + ni * 16 + (lane & 15);
      int rbase = m0 + mi * 16 + ((lane >> 4) << 3);
      float badd = (biasMode == 1) ? bias[col] : 0.f;
#pragma unroll
      for (int r = 0; r < 8; r++){
        int row = rbase + r;
        float v = acc[r] + badd;
        if (biasMode == 2) v = acc[r] + bias[((long)(row >> 7)) * N + col];
        if (doRelu) v = fmaxf(v, 0.f);
        long o = (long)row * ldc + col;
        if (Cf) Cf[o] = v;
        if (Cb) Cb[o] = (__bf16)v;
      }
    }
  }
}

// softmax over axis=1 (over i, fixed (b,j)); warp per (b,j) column.
__global__ void __launch_bounds__(256)
k_attn_softmax(const float* __restrict__ logits, float* __restrict__ w_out, __bf16* __restrict__ w_bf){
  int w    = blockIdx.x * 8 + (threadIdx.x >> 5);
  int lane = threadIdx.x & 31;
  int b = w >> 7, j = w & 127;
  float x[4];
  float m = -INFINITY;
#pragma unroll
  for (int ii = 0; ii < 4; ii++){
    int i = lane + 32 * ii;
    x[ii] = logits[(((long)(b << 7) + i) << 7) + j];
    m = fmaxf(m, x[ii]);
  }
  for (int off = 16; off; off >>= 1) m = fmaxf(m, __shfl_xor(m, off, 32));
  float s = 0.f;
#pragma unroll
  for (int ii = 0; ii < 4; ii++){ x[ii] = expf(x[ii] - m); s += x[ii]; }
  for (int off = 16; off; off >>= 1) s += __shfl_xor(s, off, 32);
  float inv = 1.f / s;
#pragma unroll
  for (int ii = 0; ii < 4; ii++){
    int i = lane + 32 * ii;
    long o = (((long)(b << 7) + i) << 7) + j;
    float v = x[ii] * inv;
    w_out[o] = v;
    w_bf[o]  = (__bf16)v;
  }
}

// gib[c] = bih[c] + (c<1024 ? bhh[c] : 0)   (fold r,z parts of bhh; n-gate bhh stays in-step)
__global__ void k_gib(const float* __restrict__ bih, const float* __restrict__ bhh, float* __restrict__ gib){
  int c = blockIdx.x * blockDim.x + threadIdx.x;
  if (c < 1536) gib[c] = bih[c] + (c < 1024 ? bhh[c] : 0.f);
}

// GRU recurrence: single persistent workgroup (32 waves = 1 WGP), h in LDS.
// wave w owns h-columns [16w,16w+16): computes gh_r/gh_z/gh_n tiles with WMMA each step.
__global__ void __launch_bounds__(1024)
k_gru(const float* __restrict__ h0, const float* __restrict__ gi, const float* __restrict__ bhh,
      const __bf16* __restrict__ Whh, float* __restrict__ ys, __bf16* __restrict__ ys_bf)
{
  __shared__ float  hs[B_ * H_];       // 8x512 f32
  __shared__ __bf16 hb[16 * 520];      // 16x512 bf16, stride 520 (bank-conflict pad)
  int tid = threadIdx.x;
  for (int i = tid; i < 16 * 520; i += 1024) hb[i] = (__bf16)0.f;
  __syncthreads();
  for (int i = tid; i < B_ * H_; i += 1024){
    float v = h0[i];
    hs[i] = v;
    hb[(i >> 9) * 520 + (i & 511)] = (__bf16)v;
  }
  __syncthreads();

  int wave = tid >> 5, lane = tid & 31;
  int c0 = wave * 16;
  int c  = c0 + (lane & 15);
  bool active = (lane < 16);
  float bn = active ? bhh[1024 + c] : 0.f;

  for (int t = 0; t < L_; t++){
    v8f aR = {}, aZ = {}, aN = {};
    for (int k0 = 0; k0 < H_; k0 += 32){
      v16bf a  = load_a_frag(hb, 520, 0, k0, lane);
      v16bf bR = load_b_frag(Whh, H_, c0,        k0, lane);
      v16bf bZ = load_b_frag(Whh, H_, 512 + c0,  k0, lane);
      v16bf bN = load_b_frag(Whh, H_, 1024 + c0, k0, lane);
      aR = WMMA_BF16(a, bR, aR);
      aZ = WMMA_BF16(a, bZ, aZ);
      aN = WMMA_BF16(a, bN, aN);
    }
    float hn[8];
    if (active){
#pragma unroll
      for (int r = 0; r < 8; r++){                 // r = batch index (M rows 8..15 discarded)
        long g = ((long)((r << 7) + t)) * 1536 + c;
        float ir = gi[g], iz = gi[g + 512], in_ = gi[g + 1024];
        float rg = sigm(ir + aR[r]);
        float z  = sigm(iz + aZ[r]);
        float n  = tanhf(in_ + rg * (aN[r] + bn));
        hn[r] = (1.f - z) * n + z * hs[(r << 9) + c];
      }
    }
    __syncthreads();                               // all waves done reading hb
    if (active){
#pragma unroll
      for (int r = 0; r < 8; r++){
        int row = (r << 7) + t;
        hs[(r << 9) + c] = hn[r];
        hb[r * 520 + c]  = (__bf16)hn[r];
        ys[(long)row * H_ + c]    = hn[r];
        ys_bf[(long)row * H_ + c] = (__bf16)hn[r];
      }
    }
    __syncthreads();                               // h updated before next step's WMMA
  }
}

// out[row] = dot(M[row,:], vec[:])  (warp per row, len=512)
__global__ void k_rowdot(const float* __restrict__ Mx, const float* __restrict__ v,
                         float* __restrict__ out){
  int w    = blockIdx.x * (blockDim.x >> 5) + (threadIdx.x >> 5);
  int lane = threadIdx.x & 31;
  const float* r = Mx + (long)w * H_;
  float s = 0.f;
  for (int k = lane; k < H_; k += 32) s += r[k] * v[k];
  for (int off = 16; off; off >>= 1) s += __shfl_down(s, off, 32);
  if (!lane) out[w] = s;
}

// copy-attention softmax over j (contiguous); warp per (b,i). Emits alphas(bf16) + row_sum.
__global__ void __launch_bounds__(256)
k_copy_softmax(const float* __restrict__ dp, const float* __restrict__ ep,
               const int* __restrict__ enc, const float* __restrict__ cbp,
               __bf16* __restrict__ al_bf, float* __restrict__ rsum){
  int w    = blockIdx.x * 8 + (threadIdx.x >> 5);
  int lane = threadIdx.x & 31;
  int b = w >> 7;
  float d = dp[w] + cbp[0];
  float x[4];
  float m = -INFINITY;
#pragma unroll
  for (int jj = 0; jj < 4; jj++){
    int j  = lane + 32 * jj;
    int bj = (b << 7) + j;
    float v = d + ep[bj] + (enc[bj] == 0 ? -1000.f : 0.f);   // PAD == 0
    x[jj] = v;
    m = fmaxf(m, v);
  }
  for (int off = 16; off; off >>= 1) m = fmaxf(m, __shfl_xor(m, off, 32));
  float s = 0.f;
#pragma unroll
  for (int jj = 0; jj < 4; jj++){ x[jj] = expf(x[jj] - m); s += x[jj]; }
  for (int off = 16; off; off >>= 1) s += __shfl_xor(s, off, 32);
  float inv = 1.f / s, s2 = 0.f;
#pragma unroll
  for (int jj = 0; jj < 4; jj++){
    int j = lane + 32 * jj;
    float v = x[jj] * inv;
    al_bf[((long)w << 7) + j] = (__bf16)v;
    s2 += v;
  }
  for (int off = 16; off; off >>= 1) s2 += __shfl_xor(s2, off, 32);
  if (!lane) rsum[w] = s2;
}

// mix[row] = sigmoid([ys,ctx,emb] . dogen_W + dogen_b)  (warp per row)
__global__ void k_mix(const float* __restrict__ ys, const float* __restrict__ ctx,
                      const float* __restrict__ emb, const float* __restrict__ dg,
                      const float* __restrict__ dgb, float* __restrict__ mix){
  int w    = blockIdx.x * (blockDim.x >> 5) + (threadIdx.x >> 5);
  int lane = threadIdx.x & 31;
  const float* a = ys  + (long)w * H_;
  const float* c = ctx + (long)w * H_;
  const float* e = emb + (long)w * H_;
  float s = 0.f;
  for (int k = lane; k < H_; k += 32)
    s += a[k] * dg[k] + c[k] * dg[H_ + k] + e[k] * dg[2 * H_ + k];
  for (int off = 16; off; off >>= 1) s += __shfl_down(s, off, 32);
  if (!lane) mix[w] = sigm(s + dgb[0]);
}

// final: in-place softmax over V on d_out + mix/copy composition; block per row
__global__ void __launch_bounds__(256)
k_final(float* __restrict__ prob, const float* __restrict__ mix,
        const float* __restrict__ rsum, const int* __restrict__ enc){
  __shared__ float red[256];
  int row = blockIdx.x, tid = threadIdx.x;
  float* p = prob + (long)row * V_;
  float m = -INFINITY;
  for (int v = tid; v < V_; v += 256) m = fmaxf(m, p[v]);
  red[tid] = m; __syncthreads();
  for (int s = 128; s; s >>= 1){ if (tid < s) red[tid] = fmaxf(red[tid], red[tid + s]); __syncthreads(); }
  m = red[0]; __syncthreads();
  float s = 0.f;
  for (int v = tid; v < V_; v += 256) s += expf(p[v] - m);
  red[tid] = s; __syncthreads();
  for (int st = 128; st; st >>= 1){ if (tid < st) red[tid] += red[tid + st]; __syncthreads(); }
  s = red[0];
  float inv = 1.f / s;
  float mx  = mix[row];
  float add = (1.f - mx) * rsum[row];
  int   tgt = enc[row];
  for (int v = tid; v < V_; v += 256){
    float g = expf(p[v] - m) * inv * mx;
    if (v == tgt) g += add;
    p[v] = g;
  }
}

// ---------------------------------------------------------------------------
extern "C" void kernel_launch(void* const* d_in, const int* in_sizes, int n_in,
                              void* d_out, int out_size, void* d_ws, size_t ws_size,
                              hipStream_t stream) {
  (void)in_sizes; (void)n_in; (void)out_size; (void)ws_size;
  const int*   input_  = (const int*)  d_in[0];
  const int*   enc_in  = (const int*)  d_in[1];
  const float* enc_hid = (const float*)d_in[2];   // (1,B,H)
  const float* enc_out = (const float*)d_in[3];   // (B,L,H)
  const float* emb_tab = (const float*)d_in[4];
  const float* attn_W  = (const float*)d_in[5];
  const float* attn_b  = (const float*)d_in[6];
  const float* comb_W  = (const float*)d_in[7];
  const float* comb_b  = (const float*)d_in[8];
  const float* Wih     = (const float*)d_in[9];
  const float* Whh     = (const float*)d_in[10];
  const float* bih     = (const float*)d_in[11];
  const float* bhh     = (const float*)d_in[12];
  const float* out_W   = (const float*)d_in[13];
  const float* out_b   = (const float*)d_in[14];
  const float* dogen_W = (const float*)d_in[15];
  const float* dogen_b = (const float*)d_in[16];
  const float* copy_W  = (const float*)d_in[17];
  const float* copy_b  = (const float*)d_in[18];

  float* prob     = (float*)d_out;                    // (B,L,V) — also logits scratch
  float* attn_out = prob + (size_t)BL_ * V_;          // (B,L,L)

  // ---- workspace carve-out (~57 MB) ----
  char* ws = (char*)d_ws;
  size_t off = 0;
  auto carve = [&](size_t bytes) -> char* {
    char* p = ws + off;
    off = (off + bytes + 255) & ~(size_t)255;
    return p;
  };
  float*  embedded   = (float*) carve((size_t)BL_ * H_ * 4);      // 2 MB
  float*  hidterm    = (float*) carve((size_t)BL_ * 4);
  float*  attn_log   = (float*) carve((size_t)BL_ * L_ * 4);      // 512 KB
  float*  gi         = (float*) carve((size_t)BL_ * 1536 * 4);    // 6 MB
  float*  ys         = (float*) carve((size_t)BL_ * H_ * 4);      // 2 MB
  float*  context    = (float*) carve((size_t)BL_ * H_ * 4);      // 2 MB
  float*  e_part     = (float*) carve((size_t)BL_ * 4);
  float*  d_part     = (float*) carve((size_t)BL_ * 4);
  float*  row_sum    = (float*) carve((size_t)BL_ * 4);
  float*  mix        = (float*) carve((size_t)BL_ * 4);
  float*  gib        = (float*) carve((size_t)1536 * 4);
  __bf16* xcat_bf    = (__bf16*)carve((size_t)BL_ * 1024 * 2);    // 2 MB  [emb | attn_applied]
  __bf16* attnW_bf   = (__bf16*)carve((size_t)L_ * 1024 * 2);     // 256 KB
  __bf16* combW_bf   = (__bf16*)carve((size_t)H_ * 1024 * 2);     // 1 MB
  __bf16* Wih_bf     = (__bf16*)carve((size_t)1536 * H_ * 2);     // 1.5 MB
  __bf16* Whh_bf     = (__bf16*)carve((size_t)1536 * H_ * 2);     // 1.5 MB
  __bf16* outW_bf    = (__bf16*)carve((size_t)V_ * H_ * 2);       // 32 MB
  __bf16* eoT_bf     = (__bf16*)carve((size_t)B_ * H_ * L_ * 2);  // 1 MB
  __bf16* wattn_bf   = (__bf16*)carve((size_t)BL_ * L_ * 2);      // 256 KB
  __bf16* alpha_bf   = (__bf16*)carve((size_t)BL_ * L_ * 2);      // 256 KB
  __bf16* x_bf       = (__bf16*)carve((size_t)BL_ * H_ * 2);      // 1 MB
  __bf16* out_bf     = (__bf16*)carve((size_t)BL_ * H_ * 2);      // 1 MB

  // 1) weight conversions to bf16
  auto cvt = [&](const float* s, __bf16* d, long n){
    int g = (int)((n + 255) / 256);
    k_cvt<<<g, 256, 0, stream>>>(s, d, n);
  };
  cvt(attn_W, attnW_bf, (long)L_ * 1024);
  cvt(comb_W, combW_bf, (long)H_ * 1024);
  cvt(Wih,    Wih_bf,   (long)1536 * H_);
  cvt(Whh,    Whh_bf,   (long)1536 * H_);
  cvt(out_W,  outW_bf,  (long)V_ * H_);

  // 2) gathers / transposes / small precomputes
  k_embed  <<<BL_, 128, 0, stream>>>(input_, emb_tab, embedded, xcat_bf);
  k_eoT    <<<2048, 256, 0, stream>>>(enc_out, eoT_bf);
  k_hidterm<<<128, 256, 0, stream>>>(enc_hid, attn_W, attn_b, hidterm);
  k_gib    <<<6, 256, 0, stream>>>(bih, bhh, gib);

  // 3) attn logits: (1024x512) @ attn_W[:, :512]^T  (+hidterm incl. attn_b)
  k_gemm<<<dim3(1, 32, 1), 256, 0, stream>>>(
      xcat_bf, 1024, 0,  attnW_bf, 1024, 0,
      attn_log, nullptr, L_, 0,  hidterm, 2,  L_, H_, 0);

  // 4) softmax over axis=1 -> attn_weights (to d_out tail) + bf16 copy
  k_attn_softmax<<<128, 256, 0, stream>>>(attn_log, attn_out, wattn_bf);

  // 5) attn_applied[b] = w[b](128x128) @ eo[b](128x512)  -> xcat[:, 512:1024] (bf16)
  k_gemm<<<dim3(2, 4, B_), 256, 0, stream>>>(
      wattn_bf, L_, (long)L_ * L_,  eoT_bf, L_, (long)H_ * L_,
      nullptr, xcat_bf + H_, 1024, (long)L_ * 1024,
      nullptr, 0,  H_, L_, 0);

  // 6) x = relu(xcat(1024x1024) @ comb_W^T + comb_b) -> bf16
  k_gemm<<<dim3(2, 32, 1), 256, 0, stream>>>(
      xcat_bf, 1024, 0,  combW_bf, 1024, 0,
      nullptr, x_bf, H_, 0,  comb_b, 1,  H_, 1024, 1);

  // 7) gi = x(1024x512) @ Wih^T + (bih + bhh[r,z])
  k_gemm<<<dim3(6, 32, 1), 256, 0, stream>>>(
      x_bf, H_, 0,  Wih_bf, H_, 0,
      gi, nullptr, 1536, 0,  gib, 1,  1536, H_, 0);

  // 8) GRU recurrence (persistent single-WGP workgroup, WMMA per step)
  k_gru<<<1, 1024, 0, stream>>>(enc_hid, gi, bhh, Whh_bf, ys, out_bf);

  // 9) generation logits: (1024x512) @ out_W^T + out_b -> d_out prob region
  k_gemm<<<dim3(125, 32, 1), 256, 0, stream>>>(
      out_bf, H_, 0,  outW_bf, H_, 0,
      prob, nullptr, V_, 0,  out_b, 1,  V_, H_, 0);

  // 10) copy attention pieces
  k_rowdot<<<128, 256, 0, stream>>>(enc_out, copy_W,       e_part);   // w_e
  k_rowdot<<<128, 256, 0, stream>>>(ys,      copy_W + H_,  d_part);   // w_d
  k_copy_softmax<<<128, 256, 0, stream>>>(d_part, e_part, enc_in, copy_b, alpha_bf, row_sum);

  // 11) context[b] = alphas[b](128x128) @ eo[b](128x512)
  k_gemm<<<dim3(2, 4, B_), 256, 0, stream>>>(
      alpha_bf, L_, (long)L_ * L_,  eoT_bf, L_, (long)H_ * L_,
      context, nullptr, H_, (long)L_ * H_,
      nullptr, 0,  H_, L_, 0);

  // 12) mix gate, then fused softmax(V) * mix + copy scatter
  k_mix  <<<128, 256, 0, stream>>>(ys, context, embedded, dogen_W, dogen_b, mix);
  k_final<<<BL_, 256, 0, stream>>>(prob, mix, row_sum, enc_in);
}
